// MV_LSTM_29703993819503
// MI455X (gfx1250) — matrix operand order repaired and checked by reading, not compiled, not run on hardware
//
#include <hip/hip_runtime.h>
#include <stdint.h>

// ---------- types ----------
typedef __bf16 bf16_t;
typedef __attribute__((ext_vector_type(16))) __bf16 v16bf;
typedef __attribute__((ext_vector_type(8)))  __bf16 v8bf;
typedef __attribute__((ext_vector_type(8)))  float  v8f;
typedef __attribute__((ext_vector_type(4)))  float  v4f;

// ---------- problem dims ----------
#define SQ   600
#define SD   48
#define NDOC 3
#define DIMD 1024
#define H    300
#define H4   1200
#define HK   320      // H padded to mult of 32 (recurrent K)
#define H2   600
#define H2P  608      // H2 padded (mult of 32, rows mult of 16)
#define SQP  608      // SQ padded to mult of 16
#define O    300
#define OP   304      // O padded to mult of 16
#define NJ   (NDOC*SD)        // 144
#define SCOLS (O*H2P)         // 182400  (s-stage col space: (o,d))
#define BCOLS (NJ*OP)         // 43776   (bil-stage col space: (n,j,o))
#define NTILE75 75            // H4/16

// ---------- workspace layout (bytes) ----------
constexpr size_t OFF_WIH_F = 0;
constexpr size_t OFF_WIH_R = OFF_WIH_F + (size_t)H4*DIMD*2;
constexpr size_t OFF_WHH_F = OFF_WIH_R + (size_t)H4*DIMD*2;
constexpr size_t OFF_WHH_R = OFF_WHH_F + (size_t)H4*HK*2;
constexpr size_t OFF_BIAS_F= OFF_WHH_R + (size_t)H4*HK*2;
constexpr size_t OFF_BIAS_R= OFF_BIAS_F + (size_t)H4*4;
constexpr size_t OFF_XQ    = OFF_BIAS_R + (size_t)H4*4;
constexpr size_t OFF_XD    = OFF_XQ + (size_t)SQP*DIMD*2;
constexpr size_t OFF_GQ    = OFF_XD + (size_t)NDOC*SD*DIMD*2;
constexpr size_t OFF_GD    = OFF_GQ + (size_t)2*SQP*H4*4;
constexpr size_t OFF_UBF   = OFF_GD + (size_t)NDOC*2*SD*H4*4;   // zero-block start
constexpr size_t OFF_VBF   = OFF_UBF + (size_t)H2P*H2P*2;
constexpr size_t OFF_S2    = OFF_VBF + (size_t)NJ*H2P*2;
constexpr size_t OFF_LINU  = OFF_S2  + (size_t)BCOLS*H2P*2;     // zero-block end
constexpr size_t OFF_LV    = OFF_LINU + (size_t)SQP*OP*4;
constexpr size_t OFF_Q     = OFF_LV  + (size_t)BCOLS*4;
constexpr size_t ZERO_BYTES = OFF_LINU - OFF_UBF;

// ---------- helpers ----------
__device__ __forceinline__ bf16_t f2bf(float f){
  uint32_t u = __builtin_bit_cast(uint32_t, f);
  uint32_t r = u + 0x7FFFu + ((u >> 16) & 1u);     // RTNE
  uint16_t h = (uint16_t)(r >> 16);
  return __builtin_bit_cast(bf16_t, h);
}
__device__ __forceinline__ float bf2f(bf16_t b){
  uint32_t u = ((uint32_t)__builtin_bit_cast(uint16_t, b)) << 16;
  return __builtin_bit_cast(float, u);
}
union ABFrag { v16bf v; struct { v8bf lo; v8bf hi; } h; };

__device__ __forceinline__ v8f wmma_bf16(v16bf a, v16bf b, v8f c){
  return __builtin_amdgcn_wmma_f32_16x16x32_bf16(false, a, false, b, (short)0, c, false, false);
}

// ---------- prep kernels ----------
__global__ void k_zero(float* p, size_t n){
  size_t i = (size_t)blockIdx.x*blockDim.x + threadIdx.x;
  size_t st = (size_t)gridDim.x*blockDim.x;
  for (; i < n; i += st) p[i] = 0.f;
}
__global__ void k_cvt(const float* s, bf16_t* d, size_t n){
  size_t i = (size_t)blockIdx.x*blockDim.x + threadIdx.x;
  if (i < n) d[i] = f2bf(s[i]);
}
__global__ void k_whh(const float* s, bf16_t* d){ // [H4][H] -> [H4][HK] zero-padded
  int i = blockIdx.x*blockDim.x + threadIdx.x;
  if (i >= H4*HK) return;
  int n = i / HK, k = i % HK;
  d[i] = (k < H) ? f2bf(s[n*H + k]) : f2bf(0.f);
}
__global__ void k_bias(const float* bihf, const float* bhhf,
                       const float* bihr, const float* bhhr,
                       float* bf, float* br){
  int i = blockIdx.x*blockDim.x + threadIdx.x;
  if (i < H4){ bf[i] = bihf[i] + bhhf[i]; br[i] = bihr[i] + bhhr[i]; }
}
__global__ void k_convx(const float* q, const float* pos, const float* neg,
                        bf16_t* xq, bf16_t* xd){
  int i = blockIdx.x*blockDim.x + threadIdx.x;
  const int NQ = SQP*DIMD, ND = NDOC*SD*DIMD;
  if (i < NQ){
    int t = i / DIMD, k = i % DIMD;
    xq[i] = (t < SQ) ? f2bf(q[(size_t)(SQ + t)*DIMD + k]) : f2bf(0.f); // batch 1
  } else if (i < NQ + ND){
    int j = i - NQ;
    int n = j / (SD*DIMD), r = j % (SD*DIMD);
    const float* src = (n == 0) ? (pos + (size_t)SD*DIMD)
                                : (neg + (size_t)((n-1)*2 + 1)*SD*DIMD); // batch 1
    xd[j] = f2bf(src[r]);
  }
}

// ---------- gate pre-activation GEMM: G = X @ Wih^T + bias ----------
__global__ void __launch_bounds__(256) k_gates(const bf16_t* __restrict__ X,
                                               const bf16_t* __restrict__ W,
                                               const float* __restrict__ bias,
                                               float* __restrict__ G, int rowTiles){
  int wave = (blockIdx.x*blockDim.x + threadIdx.x) >> 5;
  int lane = threadIdx.x & 31;
  int tasks = rowTiles * NTILE75;
  if (wave >= tasks) return;
  int rt = wave / NTILE75, ct = wave % NTILE75;
  int rowA = rt*16 + (lane & 15);
  int colB = ct*16 + (lane & 15);
  int kaoff = (lane & 16) ? 16 : 0;
  int kboff = (lane & 16) ? 16 : 0;
  const bf16_t* xrow = X + (size_t)rowA*DIMD;
  const bf16_t* wrow = W + (size_t)colB*DIMD;
  v8f acc = {0.f,0.f,0.f,0.f,0.f,0.f,0.f,0.f};
  for (int k0 = 0; k0 < DIMD; k0 += 32){
    ABFrag a;
    a.h.lo = *(const v8bf*)(xrow + k0 + kaoff);
    a.h.hi = *(const v8bf*)(xrow + k0 + kaoff + 8);
    v16bf b = *(const v16bf*)(wrow + k0 + kboff);
    acc = wmma_bf16(a.v, b, acc);
  }
  int r0 = rt*16 + ((lane & 16) ? 8 : 0);
  float bb = bias[colB];
  #pragma unroll
  for (int v = 0; v < 8; v++) G[(size_t)(r0+v)*H4 + colB] = acc[v] + bb;
}

// ---------- LSTM scan (one block per sequence x direction) ----------
__global__ void __launch_bounds__(512) k_scan(char* ws){
  int b = blockIdx.x;     // 0..7
  int dir = b & 1;
  int seq = b >> 1;       // 0=query, 1..3 docs
  const bf16_t* Whh = (const bf16_t*)(ws + (dir ? OFF_WHH_R : OFF_WHH_F));
  const float* bias_unused = nullptr; (void)bias_unused;
  const float* G; bf16_t* outBase; int S;
  int colOff = dir ? H : 0;
  if (seq == 0){
    G = (const float*)(ws + OFF_GQ) + (size_t)dir*SQP*H4;
    outBase = (bf16_t*)(ws + OFF_UBF);
    S = SQ;
  } else {
    int n = seq - 1;
    G = (const float*)(ws + OFF_GD) + (size_t)(n*2 + dir)*SD*H4;
    outBase = (bf16_t*)(ws + OFF_VBF) + (size_t)n*SD*H2P;
    S = SD;
  }
  __shared__ __align__(32) bf16_t hpad[HK];
  __shared__ float cs[H];
  __shared__ float zs[H4];
  int tid = threadIdx.x;
  if (tid < HK) hpad[tid] = f2bf(0.f);
  if (tid < H)  cs[tid] = 0.f;
  __syncthreads();
  int wave = tid >> 5, lane = tid & 31;
  int kaoff = (lane & 16) ? 16 : 0;
  int kboff = (lane & 16) ? 16 : 0;
  for (int s = 0; s < S; s++){
    int row = dir ? (S - 1 - s) : s;
    // z = G[row] + h @ Whh^T  (h replicated across A rows -> row 0 of D valid)
    for (int nt = wave; nt < NTILE75; nt += 16){
      int colB = nt*16 + (lane & 15);
      const bf16_t* wrow = Whh + (size_t)colB*HK;
      v8f acc = {0.f,0.f,0.f,0.f,0.f,0.f,0.f,0.f};
      #pragma unroll
      for (int k0 = 0; k0 < HK; k0 += 32){
        ABFrag a;
        a.h.lo = *(const v8bf*)(hpad + k0 + kaoff);
        a.h.hi = *(const v8bf*)(hpad + k0 + kaoff + 8);
        v16bf bb = *(const v16bf*)(wrow + k0 + kboff);
        acc = wmma_bf16(a.v, bb, acc);
      }
      if (lane < 16) zs[colB] = acc[0] + G[(size_t)row*H4 + colB];
    }
    __syncthreads();
    if (tid < H){
      float zi = zs[tid], zf = zs[H + tid], zg = zs[2*H + tid], zo = zs[3*H + tid];
      float c = (1.f/(1.f + __expf(-zf))) * cs[tid] + (1.f/(1.f + __expf(-zi))) * tanhf(zg);
      float h = (1.f/(1.f + __expf(-zo))) * tanhf(c);
      cs[tid] = c;
      bf16_t hb = f2bf(h);
      hpad[tid] = hb;
      outBase[(size_t)row*H2P + colOff + tid] = hb;
    }
    __syncthreads();
  }
}

// ---------- NTN stage 1: s[nj,(o,d)] = sum_h V[nj,h] * M[o,d,h]  (M read once) ----------
__global__ void __launch_bounds__(256) k_sstage(const float* __restrict__ M,
                                                const bf16_t* __restrict__ V,
                                                bf16_t* __restrict__ S2p){
  int wave = (blockIdx.x*blockDim.x + threadIdx.x) >> 5;
  int lane = threadIdx.x & 31;
  if (wave >= SCOLS/16) return;     // 11400 column tiles
  int cd0 = wave * 16;
  int o = cd0 / H2P;
  int d = (cd0 % H2P) + (lane & 15);
  bool dok = (d < H2);
  int kboff = (lane & 16) ? 16 : 0;
  int kaoff = (lane & 16) ? 16 : 0;
  const float* mrow = M + ((size_t)o*H2 + (size_t)(dok ? d : 0))*H2;
  v8f zero8 = {0.f,0.f,0.f,0.f,0.f,0.f,0.f,0.f};
  v8f acc[9];
  #pragma unroll
  for (int r = 0; r < 9; r++) acc[r] = zero8;
  for (int k0 = 0; k0 < H2P; k0 += 32){
    if (dok) __builtin_prefetch(mrow + k0 + 256, 0, 0);
    v16bf b;
    int kb = k0 + kboff;
    #pragma unroll
    for (int g = 0; g < 4; g++){
      int hh = kb + g*4;
      if (dok && (hh + 4 <= H2)){
        v4f m4 = *(const v4f*)(mrow + hh);
        #pragma unroll
        for (int e = 0; e < 4; e++) b[g*4 + e] = f2bf(m4[e]);
      } else {
        #pragma unroll
        for (int e = 0; e < 4; e++)
          b[g*4 + e] = (dok && (hh + e < H2)) ? f2bf(mrow[hh + e]) : f2bf(0.f);
      }
    }
    #pragma unroll
    for (int rt = 0; rt < 9; rt++){
      const bf16_t* vrow = V + (size_t)(rt*16 + (lane & 15))*H2P + k0 + kaoff;
      ABFrag a;
      a.h.lo = *(const v8bf*)(vrow);
      a.h.hi = *(const v8bf*)(vrow + 8);
      acc[rt] = wmma_bf16(a.v, b, acc[rt]);
    }
  }
  if (dok){
    int rbase = (lane & 16) ? 8 : 0;
    #pragma unroll
    for (int rt = 0; rt < 9; rt++){
      #pragma unroll
      for (int v = 0; v < 8; v++){
        int nj = rt*16 + rbase + v;
        S2p[((size_t)nj*OP + o)*H2P + d] = f2bf(acc[rt][v]);
      }
    }
  }
}

// ---------- linear terms: linU[i,o] = U.Wntn ; LV[c] = V.Wntn + b_bil + b_ntn ----------
__global__ void k_lin(const bf16_t* __restrict__ U, const bf16_t* __restrict__ V,
                      const float* __restrict__ Wntn, const float* __restrict__ b_bil,
                      const float* __restrict__ b_ntn, float* __restrict__ linU,
                      float* __restrict__ LV){
  int i = blockIdx.x*blockDim.x + threadIdx.x;
  const int NU = SQP*OP;
  if (i < NU){
    int r = i / OP, o = i % OP;
    float v = 0.f;
    if (o < O){
      const float* w = Wntn + (size_t)o*H2;
      const bf16_t* u = U + (size_t)r*H2P;
      for (int dd = 0; dd < H2; dd++) v += bf2f(u[dd]) * w[dd];
    }
    linU[i] = v;
  } else if (i < NU + BCOLS){
    int c = i - NU;
    int nj = c / OP, o = c % OP;
    float v = 0.f;
    if (o < O){
      v = b_bil[o] + b_ntn[o];
      const float* w = Wntn + (size_t)o*H2;
      const bf16_t* vv = V + (size_t)nj*H2P;
      for (int dd = 0; dd < H2; dd++) v += bf2f(vv[dd]) * w[dd];
    }
    LV[c] = v;
  }
}

// ---------- NTN stage 2 + relu + max-over-j pooling ----------
__global__ void __launch_bounds__(256) k_bil(const bf16_t* __restrict__ U,
                                             const bf16_t* __restrict__ S2p,
                                             const float* __restrict__ linU,
                                             const float* __restrict__ LV,
                                             float* __restrict__ Q){
  int wave = (blockIdx.x*blockDim.x + threadIdx.x) >> 5;
  int lane = threadIdx.x & 31;
  const int ITILES = SQP/16, OTILES = OP/16;        // 38, 19
  const int NTASK = NDOC * ITILES * OTILES;         // 2166
  if (wave >= NTASK) return;
  int n = wave / (ITILES*OTILES);
  int rem = wave % (ITILES*OTILES);
  int it = rem / OTILES, ot = rem % OTILES;
  int o = ot*16 + (lane & 15);
  int kaoff = (lane & 16) ? 16 : 0;
  int kboff = (lane & 16) ? 16 : 0;
  int rbase = (lane & 16) ? 8 : 0;
  const bf16_t* arow = U + (size_t)(it*16 + (lane & 15))*H2P;
  float lu[8], vmax[8];
  #pragma unroll
  for (int v = 0; v < 8; v++){
    lu[v] = linU[(size_t)(it*16 + rbase + v)*OP + o];
    vmax[v] = 0.f;                                  // relu >= 0
  }
  v8f zero8 = {0.f,0.f,0.f,0.f,0.f,0.f,0.f,0.f};
  for (int j = 0; j < SD; j++){
    int c = (n*SD + j)*OP + o;
    const bf16_t* brow = S2p + (size_t)c*H2P;
    v8f acc = zero8;
    for (int k0 = 0; k0 < H2P; k0 += 32){
      v16bf b = *(const v16bf*)(brow + k0 + kboff);
      ABFrag a;
      a.h.lo = *(const v8bf*)(arow + k0 + kaoff);
      a.h.hi = *(const v8bf*)(arow + k0 + kaoff + 8);
      acc = wmma_bf16(a.v, b, acc);
    }
    float lv = LV[c];
    #pragma unroll
    for (int v = 0; v < 8; v++){
      float val = acc[v] + lu[v] + lv;
      val = fmaxf(val, 0.f);
      vmax[v] = fmaxf(vmax[v], val);
    }
  }
  #pragma unroll
  for (int v = 0; v < 8; v++)
    Q[((size_t)n*SQP + it*16 + rbase + v)*OP + o] = vmax[v];
}

// ---------- MLP head ----------
__global__ void k_head(const float* __restrict__ Q, const float* __restrict__ W1,
                       const float* __restrict__ b1, const float* __restrict__ W2,
                       const float* __restrict__ b2, float* __restrict__ out){
  int i = blockIdx.x*blockDim.x + threadIdx.x;
  if (i >= NDOC*SQ) return;
  int n = i / SQ, t = i % SQ;
  const float* q = Q + ((size_t)n*SQP + t)*OP;
  float s = b2[0];
  for (int f = 0; f < 30; f++){
    float hv = b1[f];
    const float* w = W1 + (size_t)f*O;
    for (int o = 0; o < O; o++) hv += q[o] * w[o];
    s += W2[f] * hv;
  }
  out[i] = s;
}

// ---------- launcher ----------
extern "C" void kernel_launch(void* const* d_in, const int* in_sizes, int n_in,
                              void* d_out, int out_size, void* d_ws, size_t ws_size,
                              hipStream_t stream){
  (void)in_sizes; (void)n_in; (void)out_size; (void)ws_size;
  char* ws = (char*)d_ws;
  const float* query   = (const float*)d_in[0];
  const float* pos_doc = (const float*)d_in[1];
  const float* neg_docs= (const float*)d_in[2];
  const float* Wih_f = (const float*)d_in[5];
  const float* Whh_f = (const float*)d_in[6];
  const float* bih_f = (const float*)d_in[7];
  const float* bhh_f = (const float*)d_in[8];
  const float* Wih_r = (const float*)d_in[9];
  const float* Whh_r = (const float*)d_in[10];
  const float* bih_r = (const float*)d_in[11];
  const float* bhh_r = (const float*)d_in[12];
  const float* Mten  = (const float*)d_in[13];
  const float* b_bil = (const float*)d_in[14];
  const float* Wntn  = (const float*)d_in[15];
  const float* b_ntn = (const float*)d_in[16];
  const float* W1 = (const float*)d_in[17];
  const float* b1 = (const float*)d_in[18];
  const float* W2 = (const float*)d_in[19];
  const float* b2 = (const float*)d_in[20];

  bf16_t* wih_f = (bf16_t*)(ws + OFF_WIH_F);
  bf16_t* wih_r = (bf16_t*)(ws + OFF_WIH_R);
  bf16_t* whh_f = (bf16_t*)(ws + OFF_WHH_F);
  bf16_t* whh_r = (bf16_t*)(ws + OFF_WHH_R);
  float*  biasf = (float*)(ws + OFF_BIAS_F);
  float*  biasr = (float*)(ws + OFF_BIAS_R);
  bf16_t* xq    = (bf16_t*)(ws + OFF_XQ);
  bf16_t* xd    = (bf16_t*)(ws + OFF_XD);
  float*  gq    = (float*)(ws + OFF_GQ);
  float*  gd    = (float*)(ws + OFF_GD);
  bf16_t* ubf   = (bf16_t*)(ws + OFF_UBF);
  bf16_t* vbf   = (bf16_t*)(ws + OFF_VBF);
  bf16_t* s2    = (bf16_t*)(ws + OFF_S2);
  float*  linU  = (float*)(ws + OFF_LINU);
  float*  lv    = (float*)(ws + OFF_LV);
  float*  q     = (float*)(ws + OFF_Q);

  // 1) zero padded bf16 state (Ubf, Vbf, S2)
  k_zero<<<2048, 256, 0, stream>>>((float*)(ws + OFF_UBF), ZERO_BYTES/4);

  // 2) conversions / prep
  {
    size_t n = (size_t)H4*DIMD;
    k_cvt<<<(unsigned)((n + 255)/256), 256, 0, stream>>>(Wih_f, wih_f, n);
    k_cvt<<<(unsigned)((n + 255)/256), 256, 0, stream>>>(Wih_r, wih_r, n);
  }
  k_whh<<<(H4*HK + 255)/256, 256, 0, stream>>>(Whh_f, whh_f);
  k_whh<<<(H4*HK + 255)/256, 256, 0, stream>>>(Whh_r, whh_r);
  k_bias<<<(H4 + 255)/256, 256, 0, stream>>>(bih_f, bhh_f, bih_r, bhh_r, biasf, biasr);
  {
    int total = SQP*DIMD + NDOC*SD*DIMD;
    k_convx<<<(total + 255)/256, 256, 0, stream>>>(query, pos_doc, neg_docs, xq, xd);
  }

  // 3) gate pre-activation GEMMs (WMMA): query + 3 docs, both directions
  {
    int qtasks = (SQP/16)*NTILE75;          // 2850
    int qblk = (qtasks + 7)/8;
    k_gates<<<qblk, 256, 0, stream>>>(xq, wih_f, biasf, gq,                    SQP/16);
    k_gates<<<qblk, 256, 0, stream>>>(xq, wih_r, biasr, gq + (size_t)SQP*H4,   SQP/16);
    int dtasks = (SD/16)*NTILE75;           // 225
    int dblk = (dtasks + 7)/8;
    for (int n = 0; n < NDOC; n++){
      k_gates<<<dblk, 256, 0, stream>>>(xd + (size_t)n*SD*DIMD, wih_f, biasf,
                                        gd + (size_t)(n*2 + 0)*SD*H4, SD/16);
      k_gates<<<dblk, 256, 0, stream>>>(xd + (size_t)n*SD*DIMD, wih_r, biasr,
                                        gd + (size_t)(n*2 + 1)*SD*H4, SD/16);
    }
  }

  // 4) LSTM recurrent scans (WMMA matvec per step); 8 independent (seq,dir) blocks
  k_scan<<<8, 512, 0, stream>>>(ws);

  // 5) NTN stage 1 (streams M once, HBM-bound)
  k_sstage<<<(SCOLS/16)/8, 256, 0, stream>>>(Mten, vbf, s2);

  // 6) linear NTN terms
  {
    int total = SQP*OP + BCOLS;
    k_lin<<<(total + 255)/256, 256, 0, stream>>>(ubf, vbf, Wntn, b_bil, b_ntn, linU, lv);
  }

  // 7) NTN stage 2 + relu + max-pool over doc positions
  {
    const int NTASK = NDOC*(SQP/16)*(OP/16);  // 2166
    k_bil<<<(NTASK + 7)/8, 256, 0, stream>>>(ubf, s2, linU, lv, q);
  }

  // 8) MLP head -> d_out [3,600,1,1]
  k_head<<<(NDOC*SQ + 255)/256, 256, 0, stream>>>(q, W1, b1, W2, b2, (float*)d_out);
}
